// UstaCausalAttention_15642270892633
// MI455X (gfx1250) — matrix operand order repaired and checked by reading, not compiled, not run on hardware
//
#include <hip/hip_runtime.h>

// ---------------------------------------------------------------------------
// CDNA5 (gfx1250) causal attention, 5-pass WMMA pipeline:
//   1-3) Q/K/V = x @ W{q,k,v}        (fp32 in -> bf16 out, WMMA bf16)
//   4)   S = Q @ K^T (causal tiles only, bf16, WMMA bf16)
//   5)   P = softmax(mask(S)/sqrt(D)) row-wise, in place (VALU)
//   6)   O = P @ V (k-loop clipped by causality, fp32 NT out, WMMA bf16)
// GEMM kernel: 128x128 tile, BK=32, 256 thr = 8 wave32 (2x4), 8 WMMA/wave/step,
// double-buffered LDS with register-staged global->LDS pipeline.
// ---------------------------------------------------------------------------

typedef __attribute__((ext_vector_type(16))) __bf16          bf16x16;
typedef __attribute__((ext_vector_type(8)))  __bf16          bf16x8;
typedef __attribute__((ext_vector_type(8)))  float           f32x8;
typedef __attribute__((ext_vector_type(4)))  float           f32x4;
typedef __attribute__((ext_vector_type(8)))  unsigned short  u16x8;
typedef __attribute__((ext_vector_type(4)))  unsigned short  u16x4;

__device__ __forceinline__ unsigned short f2bf(float f) {
    unsigned int u = __builtin_bit_cast(unsigned int, f);
    u += 0x7FFFu + ((u >> 16) & 1u);           // round-to-nearest-even
    return (unsigned short)(u >> 16);
}
__device__ __forceinline__ float bf2f(unsigned short h) {
    unsigned int u = ((unsigned int)h) << 16;
    return __builtin_bit_cast(float, u);
}

constexpr int BM = 128;            // block tile M
constexpr int BN = 128;            // block tile N
constexpr int BK = 32;             // k-step == WMMA K
constexpr int LDST = BK + 8;       // padded LDS row stride (bank-conflict dodge)

// CMODE: 0 = plain, 1 = causal tile skip (scores), 2 = causal k-limit (P@V)
template <bool F32IN, bool B_TRANS, bool OUT_F32, int CMODE>
__global__ __launch_bounds__(256)
void gemm_wmma(const void* __restrict__ Ap, const void* __restrict__ Bp,
               void* __restrict__ Cp, int M, int N, int Kdim,
               int lda, int ldb, int ldc)
{
    (void)M; (void)N;
    __shared__ __align__(16) unsigned short As[2][BM * LDST];   // [m][k]
    __shared__ __align__(16) unsigned short Bs[2][BN * LDST];   // [n][k] (B^T)

    const int m0 = blockIdx.y * BM;
    const int n0 = blockIdx.x * BN;
    if (CMODE == 1 && n0 >= m0 + BM) return;          // fully masked tile
    const int Klim = (CMODE == 2) ? ((Kdim < m0 + BM) ? Kdim : (m0 + BM)) : Kdim;
    const int nsteps = Klim / BK;                     // all dims are multiples

    const int tid  = threadIdx.x;
    const int lane = tid & 31;
    const int wave = tid >> 5;
    const int wm   = wave >> 2;          // 0..1 : 64-row strip
    const int wn   = wave & 3;           // 0..3 : 32-col strip
    const int g    = lane >> 4;          // lane half (WMMA layout group)
    const int lr   = lane & 15;

    const float*          Af = (const float*)Ap;
    const unsigned short* Ab = (const unsigned short*)Ap;
    const float*          Bf = (const float*)Bp;
    const unsigned short* Bb = (const unsigned short*)Bp;

    f32x8 acc[4][2];
    #pragma unroll
    for (int i = 0; i < 4; ++i)
        #pragma unroll
        for (int j = 0; j < 2; ++j) acc[i][j] = f32x8{};

    // ---- register staging for the global->LDS pipeline ----
    f32x4 arf[4]; u16x8 arb[2];          // A tile: fp32 (4x float4) or bf16 (2x u16x8)
    f32x4 brf[4]; u16x8 brb[2];          // B tile

    auto load_tiles = [&](int k0) {
        if (F32IN) {                      // A fp32: 1024 float4s, 4 per thread
            #pragma unroll
            for (int i = 0; i < 4; ++i) {
                const int gi = tid + i * 256;
                const int r = gi >> 3, c4 = gi & 7;
                arf[i] = *(const f32x4*)&Af[(size_t)(m0 + r) * lda + k0 + c4 * 4];
            }
        } else {                          // A bf16: 512 u16x8s, 2 per thread
            #pragma unroll
            for (int i = 0; i < 2; ++i) {
                const int gi = tid + i * 256;
                const int r = gi >> 2, c8 = gi & 3;
                arb[i] = *(const u16x8*)&Ab[(size_t)(m0 + r) * lda + k0 + c8 * 8];
            }
        }
        if (B_TRANS) {                    // B = K-matrix [N][K] bf16: rows copy
            #pragma unroll
            for (int i = 0; i < 2; ++i) {
                const int gi = tid + i * 256;
                const int n = gi >> 2, c8 = gi & 3;
                brb[i] = *(const u16x8*)&Bb[(size_t)(n0 + n) * ldb + k0 + c8 * 8];
            }
        } else if (F32IN) {               // B = W [K][N] fp32
            #pragma unroll
            for (int i = 0; i < 4; ++i) {
                const int gi = tid + i * 256;
                const int k = gi >> 5, n4 = gi & 31;
                brf[i] = *(const f32x4*)&Bf[(size_t)(k0 + k) * ldb + n0 + n4 * 4];
            }
        } else {                          // B = V/P [K][N] bf16
            #pragma unroll
            for (int i = 0; i < 2; ++i) {
                const int gi = tid + i * 256;
                const int k = gi >> 4, n8 = gi & 15;
                brb[i] = *(const u16x8*)&Bb[(size_t)(k0 + k) * ldb + n0 + n8 * 8];
            }
        }
    };

    auto store_tiles = [&](int buf) {
        unsigned short* as = As[buf];
        unsigned short* bs = Bs[buf];
        if (F32IN) {
            #pragma unroll
            for (int i = 0; i < 4; ++i) {
                const int gi = tid + i * 256;
                const int r = gi >> 3, c4 = gi & 7;
                u16x4 p;
                #pragma unroll
                for (int e = 0; e < 4; ++e) p[e] = f2bf(arf[i][e]);
                *(u16x4*)&as[r * LDST + c4 * 4] = p;
            }
        } else {
            #pragma unroll
            for (int i = 0; i < 2; ++i) {
                const int gi = tid + i * 256;
                const int r = gi >> 2, c8 = gi & 3;
                *(u16x8*)&as[r * LDST + c8 * 8] = arb[i];
            }
        }
        if (B_TRANS) {                    // already [n][k]-contiguous
            #pragma unroll
            for (int i = 0; i < 2; ++i) {
                const int gi = tid + i * 256;
                const int n = gi >> 2, c8 = gi & 3;
                *(u16x8*)&bs[n * LDST + c8 * 8] = brb[i];
            }
        } else if (F32IN) {               // transpose-scatter into [n][k]
            #pragma unroll
            for (int i = 0; i < 4; ++i) {
                const int gi = tid + i * 256;
                const int k = gi >> 5, n4 = gi & 31;
                #pragma unroll
                for (int e = 0; e < 4; ++e)
                    bs[(n4 * 4 + e) * LDST + k] = f2bf(brf[i][e]);
            }
        } else {
            #pragma unroll
            for (int i = 0; i < 2; ++i) {
                const int gi = tid + i * 256;
                const int k = gi >> 4, n8 = gi & 15;
                #pragma unroll
                for (int e = 0; e < 8; ++e)
                    bs[(n8 * 8 + e) * LDST + k] = brb[i][e];
            }
        }
    };

    // ---- pipeline prologue ----
    load_tiles(0);
    store_tiles(0);
    __syncthreads();

    for (int s = 0; s < nsteps; ++s) {
        const int cur = s & 1;
        const bool has_next = (s + 1) < nsteps;
        if (has_next) load_tiles((s + 1) * BK);   // HBM loads fly over the WMMAs

        // ---- fragments per CDNA5 16-bit WMMA lane layouts ----
        // A 16x32: lanes 0-15: row=lr, k={0..7,16..23}; lanes 16-31: k={8..15,24..31}
        bf16x16 afrag[4];
        #pragma unroll
        for (int i = 0; i < 4; ++i) {
            const int row = wm * 64 + i * 16 + lr;
            const int c0  = g ? 8 : 0;
            const bf16x8 lo = *(const bf16x8*)&As[cur][row * LDST + c0];
            const bf16x8 hi = *(const bf16x8*)&As[cur][row * LDST + c0 + 16];
            afrag[i] = __builtin_shufflevector(lo, hi,
                0,1,2,3,4,5,6,7,8,9,10,11,12,13,14,15);
        }
        // B 32x16: lanes 0-15: col=lr, k={0..7,16..23}; lanes 16-31: k={8..15,24..31}
        bf16x16 bfrag[2];
        #pragma unroll
        for (int j = 0; j < 2; ++j) {
            const int col = wn * 32 + j * 16 + lr;
            const int kb  = g * 8;
            const bf16x8 lo = *(const bf16x8*)&Bs[cur][col * LDST + kb];
            const bf16x8 hi = *(const bf16x8*)&Bs[cur][col * LDST + kb + 16];
            bfrag[j] = __builtin_shufflevector(lo, hi,
                0,1,2,3,4,5,6,7,8,9,10,11,12,13,14,15);
        }
        #pragma unroll
        for (int i = 0; i < 4; ++i)
            #pragma unroll
            for (int j = 0; j < 2; ++j)
                acc[i][j] = __builtin_amdgcn_wmma_f32_16x16x32_bf16(
                    false, afrag[i], false, bfrag[j],
                    (short)0, acc[i][j], false, false);

        if (has_next) {
            store_tiles(cur ^ 1);        // other buffer: no hazard with readers
            __syncthreads();             // single barrier per k-step
        }
    }

    // ---- epilogue: C/D 16x16 f32 layout (lanes 0-15: M=r; 16-31: M=8+r) ----
    #pragma unroll
    for (int i = 0; i < 4; ++i) {
        const int rowbase = m0 + wm * 64 + i * 16 + g * 8;
        #pragma unroll
        for (int j = 0; j < 2; ++j) {
            const int col = n0 + wn * 32 + j * 16 + lr;
            #pragma unroll
            for (int r = 0; r < 8; ++r) {
                const size_t ci = (size_t)(rowbase + r) * ldc + col;
                const float v = acc[i][j][r];
                if (OUT_F32) __builtin_nontemporal_store(v, &((float*)Cp)[ci]);
                else         ((unsigned short*)Cp)[ci] = f2bf(v);
            }
        }
    }
}

// Row-wise causal softmax over bf16 scores, in place. One block per row.
__global__ __launch_bounds__(256)
void softmax_causal(unsigned short* __restrict__ P, int S)
{
    const int i = blockIdx.x;
    unsigned short* row = P + (size_t)i * S;
    const int nv = i + 1;                                  // valid keys: j <= i
    const float scale = 0.011048543456039806f;             // 1/sqrt(8192)
    const int tid = threadIdx.x;
    __shared__ float red[256];

    // pass 1: row max over valid region
    float m = -3.4e38f;
    for (int j = tid; j < nv; j += 256) m = fmaxf(m, bf2f(row[j]));
    red[tid] = m; __syncthreads();
    for (int s = 128; s > 0; s >>= 1) {
        if (tid < s) red[tid] = fmaxf(red[tid], red[tid + s]);
        __syncthreads();
    }
    m = red[0]; __syncthreads();

    // pass 2: sum of exp
    float l = 0.f;
    for (int j = tid; j < nv; j += 256) l += __expf((bf2f(row[j]) - m) * scale);
    red[tid] = l; __syncthreads();
    for (int s = 128; s > 0; s >>= 1) {
        if (tid < s) red[tid] += red[tid + s];
        __syncthreads();
    }
    l = red[0];

    // pass 3: write normalized probs; zero the masked region
    const float rinv = 1.f / l;
    for (int j = tid; j < S; j += 256) {
        const float p = (j < nv) ? __expf((bf2f(row[j]) - m) * scale) * rinv : 0.f;
        row[j] = f2bf(p);
    }
}

extern "C" void kernel_launch(void* const* d_in, const int* in_sizes, int n_in,
                              void* d_out, int out_size, void* d_ws, size_t ws_size,
                              hipStream_t stream)
{
    (void)in_sizes; (void)n_in; (void)out_size; (void)ws_size;
    const float* x  = (const float*)d_in[0];
    const float* Wq = (const float*)d_in[1];
    const float* Wk = (const float*)d_in[2];
    const float* Wv = (const float*)d_in[3];
    float* out = (float*)d_out;

    const int S = 8192, E = 1024, D = 8192;

    unsigned char* ws = (unsigned char*)d_ws;
    const size_t mat = (size_t)S * D * sizeof(unsigned short);   // 128 MiB each
    unsigned short* Qb = (unsigned short*)(ws + 0 * mat);
    unsigned short* Kb = (unsigned short*)(ws + 1 * mat);
    unsigned short* Vb = (unsigned short*)(ws + 2 * mat);
    unsigned short* Pb = (unsigned short*)(ws + 3 * mat);

    const dim3 blk(256);
    const dim3 gproj(D / BN, S / BM);     // 64 x 64
    const dim3 gfull(S / BN, S / BM);     // 64 x 64

    // Q/K/V projections: fp32 inputs converted to bf16 in the staging path
    gemm_wmma<true, false, false, 0><<<gproj, blk, 0, stream>>>(x, Wq, Qb, S, D, E, E, D, D);
    gemm_wmma<true, false, false, 0><<<gproj, blk, 0, stream>>>(x, Wk, Kb, S, D, E, E, D, D);
    gemm_wmma<true, false, false, 0><<<gproj, blk, 0, stream>>>(x, Wv, Vb, S, D, E, E, D, D);

    // S = Q @ K^T, skipping strictly-upper tiles (causal)
    gemm_wmma<false, true, false, 1><<<gfull, blk, 0, stream>>>(Qb, Kb, Pb, S, S, D, D, D, S);

    // P = softmax(mask(S)/sqrt(D)) in place
    softmax_causal<<<dim3(S), blk, 0, stream>>>(Pb, S);

    // O = P @ V with k-loop clipped to the causal band (NT fp32 output)
    gemm_wmma<false, false, true, 2><<<gfull, blk, 0, stream>>>(Pb, Vb, out, S, D, S, S, D, D);
}
